// LPKTPlus_61074434949878
// MI455X (gfx1250) — compile-verified, matrix operand order checked by reading, not run on hardware
//
#include <hip/hip_runtime.h>

typedef __bf16 v16bf __attribute__((ext_vector_type(16)));
typedef float  v8f   __attribute__((ext_vector_type(8)));
// global-address-space weight pointer: keeps loads as global_load_b128 even after laundering
typedef const __attribute__((address_space(1))) v16bf* wptr_t;

#define WMMA_BF16(ACC, A, B) \
  (ACC) = __builtin_amdgcn_wmma_f32_16x16x32_bf16(false, (A), false, (B), (short)0, (ACC), false, false)

#define SLEN  500
#define NSTEP 499

// ---- LDS layout (bytes); all regions 1KB-aligned, A-tiles are [kblk][lane(32)][16 bf16] ----
#define OFF_P0   0        // learning(t) ping   (8 KB)
#define OFF_P1   8192     // learning(t) pong   (8 KB)
#define OFF_IT   16384    // it_emb packed      (8 KB)
#define OFF_H    24576    // h packed           (8 KB)
#define OFF_X1   32768    // [q | ut | corr]    (18 KB); LG overlays first 8 KB after W1 use
#define OFF_X1C  49152
#define OFF_QN   51200    // q_next packed      (8 KB)
#define OFF_DISC 59392    // 16 f32
#define OFF_PART 59456    // 16 f32
#define SMEM_SZ  59520

union U8 { v8f v; float f[8]; };

__device__ __forceinline__ unsigned short f2bf(float x) {
  unsigned u = __float_as_uint(x);
  u = (u + 0x7FFFu + ((u >> 16) & 1u)) >> 16;   // round-to-nearest-even
  return (unsigned short)u;
}
__device__ __forceinline__ float sigmf(float x) {
  return __builtin_amdgcn_rcpf(1.f + __expf(-x));
}
__device__ __forceinline__ float tanh_f(float x) {
  float e = __expf(2.f * x);
  return 1.f - 2.f * __builtin_amdgcn_rcpf(e + 1.f);
}

// address of element (row m, k) inside a packed A-tile region (ISA 16-bit A layout, wave32)
__device__ __forceinline__ int pk_addr(int m, int k) {
  int kblk = k >> 5, kk = k & 31;
  int half = (kk & 15) >> 3;
  int base = kk - 8 * half;
  int i = (base < 8) ? base : base - 8;
  return (kblk << 10) + (m + 16 * half) * 32 + (i << 1);
}

// pack 16 consecutive k-values (c0 % 16 == 0) of row m as two 16B LDS stores
__device__ __forceinline__ void pack16(unsigned char* sm, int region, int m, int c0, const float* v) {
  int addr = region + ((c0 >> 5) << 10) + (c0 & 16);
  uint4 A, B;
  A.x = (unsigned)f2bf(v[0])  | ((unsigned)f2bf(v[1])  << 16);
  A.y = (unsigned)f2bf(v[2])  | ((unsigned)f2bf(v[3])  << 16);
  A.z = (unsigned)f2bf(v[4])  | ((unsigned)f2bf(v[5])  << 16);
  A.w = (unsigned)f2bf(v[6])  | ((unsigned)f2bf(v[7])  << 16);
  B.x = (unsigned)f2bf(v[8])  | ((unsigned)f2bf(v[9])  << 16);
  B.y = (unsigned)f2bf(v[10]) | ((unsigned)f2bf(v[11]) << 16);
  B.z = (unsigned)f2bf(v[12]) | ((unsigned)f2bf(v[13]) << 16);
  B.w = (unsigned)f2bf(v[14]) | ((unsigned)f2bf(v[15]) << 16);
  *reinterpret_cast<uint4*>(sm + addr + m * 32)        = A;
  *reinterpret_cast<uint4*>(sm + addr + (m + 16) * 32) = B;
}

// scatter a 16x16 WMMA C-fragment (this thread's 8 values) into the packed-A layout (n becomes k)
__device__ __forceinline__ void pack_frag8(unsigned char* sm, int region, int ntile, int lane, const float* vals) {
  int nl = lane & 15, hi = lane >> 4;
  int n = ntile * 16 + nl;
  int kblk = n >> 5, kk = n & 31;
  int half = (kk & 15) >> 3;
  int base = kk - 8 * half;
  int i = (base < 8) ? base : base - 8;
  int a0 = region + (kblk << 10) + (8 * hi + 16 * half) * 32 + (i << 1);
#pragma unroll
  for (int g = 0; g < 8; ++g)
    *reinterpret_cast<unsigned short*>(sm + a0 + g * 32) = f2bf(vals[g]);
}

__device__ __forceinline__ void loadrow16(const float* p, float* v) {
  const float4* q = reinterpret_cast<const float4*>(p);
  float4 a = q[0], b = q[1], c = q[2], d = q[3];
  v[0]=a.x; v[1]=a.y; v[2]=a.z;  v[3]=a.w;
  v[4]=b.x; v[5]=b.y; v[6]=b.z;  v[7]=b.w;
  v[8]=c.x; v[9]=c.y; v[10]=c.z; v[11]=c.w;
  v[12]=d.x; v[13]=d.y; v[14]=d.z; v[15]=d.w;
}

// Pack weight [N=256, K] row-major f32 -> WMMA B-operand tiles, bf16
__global__ void lpkt_pack_w(const float* __restrict__ src, unsigned short* __restrict__ dst, int K) {
  int f = blockIdx.x * 256 + threadIdx.x;
  int i    = f & 15;
  int lane = (f >> 4) & 31;
  int rest = f >> 9;
  int nkb  = K >> 5;
  int kblk  = rest % nkb;
  int ntile = rest / nkb;
  int n = ntile * 16 + (lane & 15);
  int half = lane >> 4;
  int base = (i < 8) ? i : i + 8;
  int k = (kblk << 5) + base + 8 * half;
  dst[f] = f2bf(src[n * K + k]);
}

__global__ __launch_bounds__(256, 1)
void lpkt_scan(const int* __restrict__ qseq, const int* __restrict__ itseq,
               const int* __restrict__ utseq, const float* __restrict__ corrseq,
               const float* __restrict__ Eq, const float* __restrict__ Eut,
               const float* __restrict__ Eit,
               const float* __restrict__ b1, const float* __restrict__ b2,
               const float* __restrict__ b3, const float* __restrict__ b4,
               const float* __restrict__ ba, const float* __restrict__ bd,
               const float* __restrict__ Wdisc, const float* __restrict__ bdisc,
               const v16bf* w1p_, const v16bf* w2p_,
               const v16bf* w3p_, const v16bf* w4p_,
               const v16bf* wap_, const v16bf* wdp_,
               float* __restrict__ out) {
  __shared__ __align__(32) unsigned char sm[SMEM_SZ];
  const int tid  = threadIdx.x;
  const int lane = tid & 31;
  const int w    = tid >> 5;      // wave 0..7 owns N-tiles {w, w+8}
  const int nl   = lane & 15;
  const int hi   = lane >> 4;
  float* sDisc = reinterpret_cast<float*>(sm + OFF_DISC);
  float* sPart = reinterpret_cast<float*>(sm + OFF_PART);

  // per-thread biases for its two output columns (n0 and n0+128)
  const int   n0     = w * 16 + nl;
  const float b1v[2] = {b1[n0], b1[n0 + 128]};
  const float b2v[2] = {b2[n0], b2[n0 + 128]};
  const float b3v[2] = {b3[n0], b3[n0 + 128]};
  const float b4v[2] = {b4[n0], b4[n0 + 128]};
  const float bav[2] = {ba[n0], ba[n0 + 128]};
  const float bdv[2] = {bd[n0], bd[n0 + 128]};
  const float bd0    = bdisc[0];

  // gather role: 16 threads per batch row, 16 contiguous columns each
  const int gm   = tid >> 4;
  const int c0   = (tid & 15) * 16;
  const int grow = blockIdx.x * 16 + gm;
  float wdv[16];
#pragma unroll
  for (int j = 0; j < 16; ++j) wdv[j] = Wdisc[c0 + j];

  // init: learning ping/pong = 0 (l0 = zeros), h = bf16(1.0), y[:,0] = 0
  for (int o = tid * 16; o < 16384; o += 256 * 16)
    *reinterpret_cast<uint4*>(sm + OFF_P0 + o) = make_uint4(0u, 0u, 0u, 0u);
  for (int o = tid * 16; o < 8192; o += 256 * 16)
    *reinterpret_cast<uint4*>(sm + OFF_H + o) =
        make_uint4(0x3F803F80u, 0x3F803F80u, 0x3F803F80u, 0x3F803F80u);
  U8 hfr[2];
#pragma unroll
  for (int g = 0; g < 8; ++g) { hfr[0].f[g] = 1.f; hfr[1].f[g] = 1.f; }
  if (tid < 16) out[(blockIdx.x * 16 + tid) * SLEN] = 0.f;
  __syncthreads();

  // global-AS weight pointers, re-opaqued every step so LICM cannot hoist the
  // loop-invariant B-fragment loads out of the 499-step loop (which spills)
  wptr_t w1p = (wptr_t)w1p_;
  wptr_t w2p = (wptr_t)w2p_;
  wptr_t w3p = (wptr_t)w3p_;
  wptr_t w4p = (wptr_t)w4p_;
  wptr_t wap = (wptr_t)wap_;
  wptr_t wdp = (wptr_t)wdp_;

#define BIDX(TILE, NK, KB) (((TILE) * (NK) + (KB)) * 32 + lane)
#define LDSA(OFF) (*reinterpret_cast<const v16bf*>(sm + (OFF) + lane * 32))

  U8 lg[2];
  for (int t = 0; t < NSTEP; ++t) {
    asm volatile("" : "+s"(w1p), "+s"(w2p), "+s"(w3p), "+s"(w4p), "+s"(wap), "+s"(wdp));
    const int loff  = OFF_P0 + ((t & 1) << 13);        // learning(t)
    const int lpoff = OFF_P0 + (((t & 1) ^ 1) << 13);  // learning(t-1)

    // ---- phase A: gather embeddings, pack A-tiles, disc dot ----
    {
      const int bidx = grow * SLEN + t;
      const int qt = qseq[bidx], it = itseq[bidx], ut = utseq[bidx], qn = qseq[bidx + 1];
      const float cv = corrseq[bidx];
      float v[16];
      loadrow16(Eq  + qt * 256 + c0, v); pack16(sm, OFF_X1,        gm, c0, v);
      loadrow16(Eut + ut * 256 + c0, v); pack16(sm, OFF_X1 + 8192, gm, c0, v);
      loadrow16(Eit + it * 256 + c0, v); pack16(sm, OFF_IT,        gm, c0, v);
      loadrow16(Eq  + qn * 256 + c0, v); pack16(sm, OFF_QN,        gm, c0, v);
      float ds = 0.f;
#pragma unroll
      for (int j = 0; j < 16; ++j) ds += v[j] * wdv[j];
#pragma unroll
      for (int off = 1; off < 16; off <<= 1) ds += __shfl_xor(ds, off, 32);
      if (lane == 0)  sDisc[2 * w]     = ds;
      if (lane == 16) sDisc[2 * w + 1] = ds;
      const unsigned short cb = f2bf(cv);
#pragma unroll
      for (int j = 0; j < 4; ++j) {
        int k = (tid & 15) * 4 + j;    // 64 correctness-dim copies per row
        *reinterpret_cast<unsigned short*>(sm + OFF_X1C + pk_addr(gm, k)) = cb;
      }
    }
    __syncthreads();

    if (tid < 16) {
      sDisc[tid] = 10.f * sigmf(sDisc[tid] + bd0);
      sPart[tid] = 0.f;
    }

    // ---- learning(t) = [q|ut|corr] @ W1^T + b1   (K = 576), 2-deep pipeline ----
    {
      U8 acc[2];
#pragma unroll
      for (int g = 0; g < 8; ++g) { acc[0].f[g] = 0.f; acc[1].f[g] = 0.f; }
      v16bf a  = LDSA(OFF_X1);
      v16bf p0 = w1p[BIDX(w, 18, 0)];
      v16bf p1 = w1p[BIDX(w + 8, 18, 0)];
#pragma unroll
      for (int kb = 0; kb < 18; ++kb) {
        v16bf an = a, q0 = p0, q1 = p1;
        if (kb + 1 < 18) {
          an = LDSA(OFF_X1 + ((kb + 1) << 10));
          q0 = w1p[BIDX(w, 18, kb + 1)];
          q1 = w1p[BIDX(w + 8, 18, kb + 1)];
        }
        WMMA_BF16(acc[0].v, a, p0);
        WMMA_BF16(acc[1].v, a, p1);
        a = an; p0 = q0; p1 = q1;
      }
#pragma unroll
      for (int g = 0; g < 8; ++g) { acc[0].f[g] += b1v[0]; acc[1].f[g] += b1v[1]; }
      pack_frag8(sm, loff, w,     lane, acc[0].f);
      pack_frag8(sm, loff, w + 8, lane, acc[1].f);
    }
    __syncthreads();

    // ---- gates: x = [l(t-1)|it|l(t)|h]  (K = 1024) through W2 (tanh) and W3 (sigm) ----
    {
      U8 g2[2], g3[2];
#pragma unroll
      for (int g = 0; g < 8; ++g) { g2[0].f[g]=0.f; g2[1].f[g]=0.f; g3[0].f[g]=0.f; g3[1].f[g]=0.f; }
#define XOFF(KB) ((KB) < 8  ? lpoff  + ((KB) << 10)        \
                : (KB) < 16 ? OFF_IT + (((KB) - 8) << 10)  \
                : (KB) < 24 ? loff   + (((KB) - 16) << 10) \
                            : OFF_H  + (((KB) - 24) << 10))
      v16bf a   = LDSA(XOFF(0));
      v16bf p20 = w2p[BIDX(w, 32, 0)];
      v16bf p21 = w2p[BIDX(w + 8, 32, 0)];
      v16bf p30 = w3p[BIDX(w, 32, 0)];
      v16bf p31 = w3p[BIDX(w + 8, 32, 0)];
#pragma unroll
      for (int kb = 0; kb < 32; ++kb) {
        v16bf an = a, q20 = p20, q21 = p21, q30 = p30, q31 = p31;
        if (kb + 1 < 32) {
          an  = LDSA(XOFF(kb + 1));
          q20 = w2p[BIDX(w, 32, kb + 1)];
          q21 = w2p[BIDX(w + 8, 32, kb + 1)];
          q30 = w3p[BIDX(w, 32, kb + 1)];
          q31 = w3p[BIDX(w + 8, 32, kb + 1)];
        }
        WMMA_BF16(g2[0].v, a, p20);
        WMMA_BF16(g2[1].v, a, p21);
        WMMA_BF16(g3[0].v, a, p30);
        WMMA_BF16(g3[1].v, a, p31);
        a = an; p20 = q20; p21 = q21; p30 = q30; p31 = q31;
      }
#undef XOFF
#pragma unroll
      for (int ti = 0; ti < 2; ++ti)
#pragma unroll
        for (int g = 0; g < 8; ++g) {
          float gain = tanh_f(g2[ti].f[g] + b2v[ti]);
          float gl   = sigmf(g3[ti].f[g] + b3v[ti]);
          lg[ti].f[g] = gl * (gain + 1.f) * 0.5f;
        }
    }
    pack_frag8(sm, OFF_X1, w,     lane, lg[0].f);   // LG overlays q region (W1 consumers done)
    pack_frag8(sm, OFF_X1, w + 8, lane, lg[1].f);
    __syncthreads();

    // ---- gamma_f = sigm([h|LG|it] @ W4^T + b4);  h = LG + gamma_f * h_pre ----
    {
      U8 g4[2];
#pragma unroll
      for (int g = 0; g < 8; ++g) { g4[0].f[g] = 0.f; g4[1].f[g] = 0.f; }
#define XOFF(KB) ((KB) < 8  ? OFF_H  + ((KB) << 10)        \
                : (KB) < 16 ? OFF_X1 + (((KB) - 8) << 10)  \
                            : OFF_IT + (((KB) - 16) << 10))
      v16bf a  = LDSA(XOFF(0));
      v16bf p0 = w4p[BIDX(w, 24, 0)];
      v16bf p1 = w4p[BIDX(w + 8, 24, 0)];
#pragma unroll
      for (int kb = 0; kb < 24; ++kb) {
        v16bf an = a, q0 = p0, q1 = p1;
        if (kb + 1 < 24) {
          an = LDSA(XOFF(kb + 1));
          q0 = w4p[BIDX(w, 24, kb + 1)];
          q1 = w4p[BIDX(w + 8, 24, kb + 1)];
        }
        WMMA_BF16(g4[0].v, a, p0);
        WMMA_BF16(g4[1].v, a, p1);
        a = an; p0 = q0; p1 = q1;
      }
#undef XOFF
      __syncthreads();               // all W4 reads of OFF_H done before overwrite
#pragma unroll
      for (int ti = 0; ti < 2; ++ti)
#pragma unroll
        for (int g = 0; g < 8; ++g) {
          float gf = sigmf(g4[ti].f[g] + b4v[ti]);
          hfr[ti].f[g] = lg[ti].f[g] + gf * hfr[ti].f[g];
        }
      pack_frag8(sm, OFF_H, w,     lane, hfr[0].f);
      pack_frag8(sm, OFF_H, w + 8, lane, hfr[1].f);
    }
    __syncthreads();

    // ---- prediction: ability = sigm(h@Wa^T+ba), diff = sigm(qn@Wd^T+bd), row-reduce ----
    {
      U8 aa[2], dd[2];
#pragma unroll
      for (int g = 0; g < 8; ++g) { aa[0].f[g]=0.f; aa[1].f[g]=0.f; dd[0].f[g]=0.f; dd[1].f[g]=0.f; }
      v16bf ah  = LDSA(OFF_H);
      v16bf aq  = LDSA(OFF_QN);
      v16bf pa0 = wap[BIDX(w, 8, 0)];
      v16bf pa1 = wap[BIDX(w + 8, 8, 0)];
      v16bf pd0 = wdp[BIDX(w, 8, 0)];
      v16bf pd1 = wdp[BIDX(w + 8, 8, 0)];
#pragma unroll
      for (int kb = 0; kb < 8; ++kb) {
        v16bf ahn = ah, aqn = aq, qa0 = pa0, qa1 = pa1, qd0 = pd0, qd1 = pd1;
        if (kb + 1 < 8) {
          ahn = LDSA(OFF_H  + ((kb + 1) << 10));
          aqn = LDSA(OFF_QN + ((kb + 1) << 10));
          qa0 = wap[BIDX(w, 8, kb + 1)];
          qa1 = wap[BIDX(w + 8, 8, kb + 1)];
          qd0 = wdp[BIDX(w, 8, kb + 1)];
          qd1 = wdp[BIDX(w + 8, 8, kb + 1)];
        }
        WMMA_BF16(aa[0].v, ah, pa0);
        WMMA_BF16(aa[1].v, ah, pa1);
        WMMA_BF16(dd[0].v, aq, pd0);
        WMMA_BF16(dd[1].v, aq, pd1);
        ah = ahn; aq = aqn; pa0 = qa0; pa1 = qa1; pd0 = qd0; pd1 = qd1;
      }
      float s[8];
#pragma unroll
      for (int g = 0; g < 8; ++g) {
        float a0v = sigmf(aa[0].f[g] + bav[0]);
        float d0v = sigmf(dd[0].f[g] + bdv[0]);
        float a1v = sigmf(aa[1].f[g] + bav[1]);
        float d1v = sigmf(dd[1].f[g] + bdv[1]);
        s[g] = (a0v - d0v) * d0v + (a1v - d1v) * d1v;
      }
#pragma unroll
      for (int off = 1; off < 16; off <<= 1)
#pragma unroll
        for (int g = 0; g < 8; ++g) s[g] += __shfl_xor(s[g], off, 32);
      if (nl == 0) {
#pragma unroll
        for (int g = 0; g < 8; ++g) atomicAdd(&sPart[g + 8 * hi], s[g]);
      }
    }
    __syncthreads();
    if (tid < 16)
      out[(blockIdx.x * 16 + tid) * SLEN + t + 1] = sigmf(sDisc[tid] * sPart[tid]);
    __syncthreads();
  }
#undef BIDX
#undef LDSA
}

extern "C" void kernel_launch(void* const* d_in, const int* in_sizes, int n_in,
                              void* d_out, int out_size, void* d_ws, size_t ws_size,
                              hipStream_t stream) {
  (void)in_sizes; (void)n_in; (void)out_size; (void)ws_size;
  const int*   qseq  = (const int*)d_in[0];
  const int*   itseq = (const int*)d_in[1];
  const int*   utseq = (const int*)d_in[2];
  const float* corr  = (const float*)d_in[3];
  const float* Eq    = (const float*)d_in[4];
  const float* Eut   = (const float*)d_in[5];
  const float* Eit   = (const float*)d_in[6];
  const float* W1    = (const float*)d_in[7];
  const float* b1    = (const float*)d_in[8];
  const float* W2    = (const float*)d_in[9];
  const float* b2    = (const float*)d_in[10];
  const float* W3    = (const float*)d_in[11];
  const float* b3    = (const float*)d_in[12];
  const float* W4    = (const float*)d_in[13];
  const float* b4    = (const float*)d_in[14];
  const float* Wa    = (const float*)d_in[15];
  const float* ba    = (const float*)d_in[16];
  const float* Wd    = (const float*)d_in[17];
  const float* bd    = (const float*)d_in[18];
  const float* Wdisc = (const float*)d_in[19];
  const float* bdisc = (const float*)d_in[20];

  // workspace: packed bf16 weights (u16 element offsets), ~2.0 MB total
  unsigned short* ws  = (unsigned short*)d_ws;
  unsigned short* w1p = ws;              // 256*576
  unsigned short* w2p = ws + 147456;     // 256*1024
  unsigned short* w3p = ws + 409600;     // 256*1024
  unsigned short* w4p = ws + 671744;     // 256*768
  unsigned short* wap = ws + 868352;     // 256*256
  unsigned short* wdp = ws + 933888;     // 256*256

  lpkt_pack_w<<<576,  256, 0, stream>>>(W1, w1p, 576);
  lpkt_pack_w<<<1024, 256, 0, stream>>>(W2, w2p, 1024);
  lpkt_pack_w<<<1024, 256, 0, stream>>>(W3, w3p, 1024);
  lpkt_pack_w<<<768,  256, 0, stream>>>(W4, w4p, 768);
  lpkt_pack_w<<<256,  256, 0, stream>>>(Wa, wap, 256);
  lpkt_pack_w<<<256,  256, 0, stream>>>(Wd, wdp, 256);

  lpkt_scan<<<16, 256, 0, stream>>>(qseq, itseq, utseq, corr, Eq, Eut, Eit,
                                    b1, b2, b3, b4, ba, bd, Wdisc, bdisc,
                                    (const v16bf*)w1p, (const v16bf*)w2p,
                                    (const v16bf*)w3p, (const v16bf*)w4p,
                                    (const v16bf*)wap, (const v16bf*)wdp,
                                    (float*)d_out);
}
